// SpMiddlePillarEncoder18_12343736008743
// MI455X (gfx1250) — matrix operand order repaired and checked by reading, not compiled, not run on hardware
//
#include <hip/hip_runtime.h>

typedef _Float16 f16;
typedef __attribute__((ext_vector_type(8)))  _Float16 v8h;
typedef __attribute__((ext_vector_type(16))) _Float16 v16h;
typedef __attribute__((ext_vector_type(8)))  float    v8f;

// ---------- prep: x (NCHW f32) * mask -> NHWC f16 ; mask -> f32 map ----------
__global__ void k_prep(const float* __restrict__ x, const int* __restrict__ mask,
                       f16* __restrict__ X, float* __restrict__ M,
                       int Bn, int C, int H, int W) {
  int i = blockIdx.x * blockDim.x + threadIdx.x;
  int total = Bn * H * W;
  if (i >= total) return;
  int w = i % W;
  int h = (i / W) % H;
  int n = i / (W * H);
  float m = (mask[i] != 0) ? 1.0f : 0.0f;
  M[i] = m;
  size_t plane = (size_t)H * W;
  const float* src = x + ((size_t)n * C) * plane + (size_t)h * W + w;
  f16* dst = X + (size_t)i * C;
  for (int c = 0; c < C; ++c)
    dst[c] = (f16)(src[(size_t)c * plane] * m);
}

// ---------- weight convert: OIHW f32 -> [tap][Cout][Cin] f16 ----------
__global__ void k_wcvt(const float* __restrict__ w, f16* __restrict__ wb,
                       int Cout, int Cin) {
  int i = blockIdx.x * blockDim.x + threadIdx.x;
  int total = Cout * Cin * 9;
  if (i >= total) return;
  int tap = i % 9;
  int ci  = (i / 9) % Cin;
  int co  = i / (9 * Cin);
  wb[((size_t)tap * Cout + co) * Cin + ci] = (f16)w[(size_t)(co * Cin + ci) * 9 + tap];
}

// ---------- mask downsample: stride-2 3x3 OR-window ----------
__global__ void k_maskdown(const float* __restrict__ mp, float* __restrict__ mn,
                           int Bn, int Hp, int Wp, int Ho, int Wo) {
  int i = blockIdx.x * blockDim.x + threadIdx.x;
  int total = Bn * Ho * Wo;
  if (i >= total) return;
  int ox = i % Wo;
  int oy = (i / Wo) % Ho;
  int n  = i / (Wo * Ho);
  float any = 0.0f;
  for (int ky = 0; ky < 3; ++ky) {
    int iy = 2 * oy + ky - 1;
    if (iy < 0 || iy >= Hp) continue;
    for (int kx = 0; kx < 3; ++kx) {
      int ix = 2 * ox + kx - 1;
      if (ix < 0 || ix >= Wp) continue;
      if (mp[((size_t)n * Hp + iy) * Wp + ix] > 0.0f) any = 1.0f;
    }
  }
  mn[i] = any;
}

// ---------- WMMA MAC core: 9 taps x (CIN/32) K-chunks, dual N-tiles ----------
template <int CIN, bool CHECK>
__device__ __forceinline__ void conv_core(
    const f16* __restrict__ in, const f16* __restrict__ wb,
    int n, int oy, int oxA, int co0, int hi,
    int Hi, int Wi, int Cout, int stride,
    v8f& acc0, v8f& acc1) {
  for (int ky = 0; ky < 3; ++ky) {
    int iy = oy * stride + ky - 1;
    if (iy < 0 || iy >= Hi) continue;           // wave-uniform
    const f16* aplane = in + ((size_t)n * Hi + iy) * (size_t)Wi * CIN;
    for (int kx = 0; kx < 3; ++kx) {
      int ix = oxA * stride + kx - 1;
      bool okx = !CHECK || ((ix >= 0) && (ix < Wi));
      const f16* arow = aplane + (size_t)(okx ? ix : 0) * CIN;
      int tap = ky * 3 + kx;
      const f16* brow0 = wb + ((size_t)tap * Cout + co0) * CIN + hi * 16;
      const f16* brow1 = brow0 + (size_t)16 * CIN;   // co0+16 column group
#pragma unroll
      for (int c0 = 0; c0 < CIN; c0 += 32) {
        v8h a0 = {}, a1 = {};
        if (okx) {                                   // zero-pad outside image
          a0 = *(const v8h*)(arow + c0 + hi * 16);       // K = hi*16 .. +7
          a1 = *(const v8h*)(arow + c0 + hi * 16 + 8);   // K = hi*16+8 ..
        }
        v16h A = __builtin_shufflevector(a0, a1,
            0, 1, 2, 3, 4, 5, 6, 7, 8, 9, 10, 11, 12, 13, 14, 15);
        v16h B0 = *(const v16h*)(brow0 + c0);            // K = hi*16 .. +15
        v16h B1 = *(const v16h*)(brow1 + c0);
        acc0 = __builtin_amdgcn_wmma_f32_16x16x32_f16(
            false, A, false, B0, (short)0, acc0, false, false);
        acc1 = __builtin_amdgcn_wmma_f32_16x16x32_f16(
            false, A, false, B1, (short)0, acc1, false, false);
      }
    }
  }
}

// ---------- implicit-GEMM 3x3 conv: one wave = 16(spatial) x 32(Cout) ----------
// in:  NHWC f16 [Bn][Hi][Wi][CIN]   wb: [9][Cout][CIN] f16
// out: NHWC f16 [Bn][Ho][Wo][Cout]  out_nchw (opt): f32 NCHW stage output
// epilogue: v = acc*s + (bias*s+t) ; v*=mask ; v+=resid(own elem) ; relu
template <int CIN>
__global__ void __launch_bounds__(256) k_conv_wmma(
    const f16* __restrict__ in, const f16* __restrict__ wb,
    const float* __restrict__ bias, const float* __restrict__ sc,
    const float* __restrict__ sh, const float* __restrict__ mask,
    const f16* __restrict__ resid, f16* __restrict__ out,
    float* __restrict__ out_nchw,
    int Bn, int Hi, int Wi, int Ho, int Wo, int Cout,
    int lgNTx, int lgNTc, int lgHo, int stride) {
  int wave = blockIdx.x * (blockDim.x >> 5) + (threadIdx.x >> 5);
  int nTx = 1 << lgNTx, nTc = 1 << lgNTc;
  int total = Bn << (lgHo + lgNTx + lgNTc);
  if (wave >= total) return;                   // wave-uniform exit
  int ct = wave & (nTc - 1); int t = wave >> lgNTc;
  int xt = t & (nTx - 1);    t >>= lgNTx;
  int oy = t & ((1 << lgHo) - 1);
  int n  = t >> lgHo;

  int lane = threadIdx.x & 31;
  int lo = lane & 15, hi = lane >> 4;
  int oxA = (xt << 4) + lo;                    // A-fragment row (spatial M)
  int co0 = (ct << 5) + lo;                    // first of two N columns
  int co1 = co0 + 16;

  v8f acc0 = {}, acc1 = {};
  // interior tiles never touch the x-padding -> skip all per-lane checks
  bool interior = (xt > 0) && ((stride == 2) || (xt < nTx - 1));
  if (interior)
    conv_core<CIN, false>(in, wb, n, oy, oxA, co0, hi, Hi, Wi, Cout, stride, acc0, acc1);
  else
    conv_core<CIN, true>(in, wb, n, oy, oxA, co0, hi, Hi, Wi, Cout, stride, acc0, acc1);

  // fold bias into BN shift:  (acc+b)*s+t = acc*s + (b*s+t)
  float s0 = sc[co0], s1 = sc[co1];
  float c0f = sh[co0], c1f = sh[co1];
  if (bias) { c0f += bias[co0] * s0; c1f += bias[co1] * s1; }

  size_t rowbase = ((size_t)n * Ho + oy) * Wo + (xt << 4) + (hi << 3);
  v8f mv = *(const v8f*)(mask + rowbase);      // 8 consecutive mask values
#pragma unroll
  for (int r = 0; r < 8; ++r) {
    size_t pos = rowbase + r;                  // D: M = r + 8*hi, N = lo
    int ox = (xt << 4) + (hi << 3) + r;
    float mr = mv[r];
    float v0 = (acc0[r] * s0 + c0f) * mr;
    float v1 = (acc1[r] * s1 + c1f) * mr;
    if (resid) {                               // read own elem, then write
      v0 += (float)resid[pos * Cout + co0];
      v1 += (float)resid[pos * Cout + co1];
    }
    v0 = v0 > 0.0f ? v0 : 0.0f;
    v1 = v1 > 0.0f ? v1 : 0.0f;
    out[pos * Cout + co0] = (f16)v0;
    out[pos * Cout + co1] = (f16)v1;
    if (out_nchw) {
      out_nchw[(((size_t)n * Cout + co0) * Ho + oy) * Wo + ox] = v0;
      out_nchw[(((size_t)n * Cout + co1) * Ho + oy) * Wo + ox] = v1;
    }
  }
}

extern "C" void kernel_launch(void* const* d_in, const int* in_sizes, int n_in,
                              void* d_out, int out_size, void* d_ws, size_t ws_size,
                              hipStream_t stream) {
  (void)in_sizes; (void)n_in; (void)out_size; (void)ws_size;
  const int Bn = 2;
  const float* x  = (const float*)d_in[0];
  const int* mask = (const int*)d_in[1];
  float* outp = (float*)d_out;

  char* ws = (char*)d_ws;
  size_t off = 0;
  auto alloc = [&](size_t bytes) -> void* {
    off = (off + 255) & ~(size_t)255;
    void* p = ws + off;
    off += bytes;
    return p;
  };
  // Two ping-pong activation arenas sized for stage 1 (2*512*512*32 f16)
  f16* R0 = (f16*)alloc(2ull * 512 * 512 * 32 * 2);
  f16* R1 = (f16*)alloc(2ull * 512 * 512 * 32 * 2);
  float* M1 = (float*)alloc(2ull * 512 * 512 * 4);
  float* M2 = (float*)alloc(2ull * 256 * 256 * 4);
  float* M3 = (float*)alloc(2ull * 128 * 128 * 4);
  float* M4 = (float*)alloc(2ull * 64 * 64 * 4);

  // 20 convs: {input-index of w, Cout, Cin}; insertion-order flattening of params
  const int wm[20][3] = {
    {2, 32, 32}, {6, 32, 32}, {10, 32, 32},          // b1a w0,w1,w2
    {14, 32, 32}, {18, 32, 32},                      // b1b w1,w2
    {22, 64, 32},                                    // d2
    {25, 64, 64}, {29, 64, 64}, {33, 64, 64}, {37, 64, 64},      // b2a,b2b
    {41, 128, 64},                                   // d3
    {44, 128, 128}, {48, 128, 128}, {52, 128, 128}, {56, 128, 128}, // b3a,b3b
    {60, 256, 128},                                  // d4
    {63, 256, 256}, {67, 256, 256}, {71, 256, 256}, {75, 256, 256}  // b4a,b4b
  };
  f16* WB[20];
  for (int i = 0; i < 20; ++i) {
    int Cout = wm[i][1], Cin = wm[i][2];
    WB[i] = (f16*)alloc((size_t)9 * Cout * Cin * 2);
    int tot = Cout * Cin * 9;
    k_wcvt<<<(tot + 255) / 256, 256, 0, stream>>>((const float*)d_in[wm[i][0]],
                                                  WB[i], Cout, Cin);
  }

  { int tot = Bn * 512 * 512;
    k_prep<<<(tot + 255) / 256, 256, 0, stream>>>(x, mask, R0, M1, Bn, 32, 512, 512); }

  auto F = [&](int i) { return (const float*)d_in[i]; };
  auto conv = [&](const f16* in, f16* w, const float* b, const float* s,
                  const float* t, const float* mk, const f16* rs, f16* ob,
                  float* o2, int Hi, int Wi, int Ci, int Ho, int Wo, int Co,
                  int stride) {
    int nTx = Wo >> 4, nTc = Co >> 5;
    int waves = Bn * Ho * nTx * nTc;
    int blocks = (waves + 7) / 8;
    int lgNTx = __builtin_ctz(nTx), lgNTc = __builtin_ctz(nTc),
        lgHo = __builtin_ctz(Ho);
    switch (Ci) {
      case 32:
        k_conv_wmma<32><<<blocks, 256, 0, stream>>>(in, w, b, s, t, mk, rs, ob, o2,
            Bn, Hi, Wi, Ho, Wo, Co, lgNTx, lgNTc, lgHo, stride);
        break;
      case 64:
        k_conv_wmma<64><<<blocks, 256, 0, stream>>>(in, w, b, s, t, mk, rs, ob, o2,
            Bn, Hi, Wi, Ho, Wo, Co, lgNTx, lgNTc, lgHo, stride);
        break;
      case 128:
        k_conv_wmma<128><<<blocks, 256, 0, stream>>>(in, w, b, s, t, mk, rs, ob, o2,
            Bn, Hi, Wi, Ho, Wo, Co, lgNTx, lgNTc, lgHo, stride);
        break;
      default:
        k_conv_wmma<256><<<blocks, 256, 0, stream>>>(in, w, b, s, t, mk, rs, ob, o2,
            Bn, Hi, Wi, Ho, Wo, Co, lgNTx, lgNTc, lgHo, stride);
        break;
    }
  };

  // ---- Stage 1: 512x512, C=32 ----
  conv(R0, WB[0], F(3),  F(4),  F(5),  M1, nullptr, R1, nullptr, 512,512,32, 512,512,32, 1); // b1a.conv0
  conv(R1, WB[1], F(7),  F(8),  F(9),  M1, nullptr, R0, nullptr, 512,512,32, 512,512,32, 1); // b1a.conv1
  conv(R0, WB[2], F(11), F(12), F(13), M1, R1,      R1, nullptr, 512,512,32, 512,512,32, 1); // b1a.conv2 (+resid)
  conv(R1, WB[3], F(15), F(16), F(17), M1, nullptr, R0, nullptr, 512,512,32, 512,512,32, 1); // b1b.conv1
  conv(R0, WB[4], F(19), F(20), F(21), M1, R1,      R1, nullptr, 512,512,32, 512,512,32, 1); // b1b.conv2

  // ---- d2 + Stage 2: 256x256, C=64 ----
  { int tot = Bn * 256 * 256;
    k_maskdown<<<(tot + 255) / 256, 256, 0, stream>>>(M1, M2, Bn, 512, 512, 256, 256); }
  conv(R1, WB[5], nullptr, F(23), F(24), M2, nullptr, R0, nullptr, 512,512,32, 256,256,64, 2); // d2
  conv(R0, WB[6], F(26), F(27), F(28), M2, nullptr, R1, nullptr, 256,256,64, 256,256,64, 1);   // b2a.conv1
  conv(R1, WB[7], F(30), F(31), F(32), M2, R0,      R0, nullptr, 256,256,64, 256,256,64, 1);   // b2a.conv2
  conv(R0, WB[8], F(34), F(35), F(36), M2, nullptr, R1, nullptr, 256,256,64, 256,256,64, 1);   // b2b.conv1
  conv(R1, WB[9], F(38), F(39), F(40), M2, R0,      R0, outp,    256,256,64, 256,256,64, 1);   // b2b.conv2 -> x2

  // ---- d3 + Stage 3: 128x128, C=128 ----
  { int tot = Bn * 128 * 128;
    k_maskdown<<<(tot + 255) / 256, 256, 0, stream>>>(M2, M3, Bn, 256, 256, 128, 128); }
  conv(R0, WB[10], nullptr, F(42), F(43), M3, nullptr, R1, nullptr, 256,256,64, 128,128,128, 2); // d3
  conv(R1, WB[11], F(45), F(46), F(47), M3, nullptr, R0, nullptr, 128,128,128, 128,128,128, 1);  // b3a.conv1
  conv(R0, WB[12], F(49), F(50), F(51), M3, R1,      R1, nullptr, 128,128,128, 128,128,128, 1);  // b3a.conv2
  conv(R1, WB[13], F(53), F(54), F(55), M3, nullptr, R0, nullptr, 128,128,128, 128,128,128, 1);  // b3b.conv1
  conv(R0, WB[14], F(57), F(58), F(59), M3, R1,      R1, outp + 8388608,
       128,128,128, 128,128,128, 1);                                                             // b3b.conv2 -> x3

  // ---- d4 + Stage 4: 64x64, C=256 ----
  { int tot = Bn * 64 * 64;
    k_maskdown<<<(tot + 255) / 256, 256, 0, stream>>>(M3, M4, Bn, 128, 128, 64, 64); }
  conv(R1, WB[15], nullptr, F(61), F(62), M4, nullptr, R0, nullptr, 128,128,128, 64,64,256, 2);  // d4
  conv(R0, WB[16], F(64), F(65), F(66), M4, nullptr, R1, nullptr, 64,64,256, 64,64,256, 1);      // b4a.conv1
  conv(R1, WB[17], F(68), F(69), F(70), M4, R0,      R0, nullptr, 64,64,256, 64,64,256, 1);      // b4a.conv2
  conv(R0, WB[18], F(72), F(73), F(74), M4, nullptr, R1, nullptr, 64,64,256, 64,64,256, 1);      // b4b.conv1
  conv(R1, WB[19], F(76), F(77), F(78), M4, R0,      R0, outp + 12582912,
       64,64,256, 64,64,256, 1);                                                                 // b4b.conv2 -> x4
}
